// NATLayer_30124900614217
// MI455X (gfx1250) — matrix operand order repaired and checked by reading, not compiled, not run on hardware
//
#include <hip/hip_runtime.h>
#include <math.h>
#include <stdint.h>

// ---------------------------------------------------------------- constants
#define NB     2
#define HH     96
#define WW     96
#define CC     256
#define HEADS  8
#define PP     25
#define DHH    32
#define HIDN   1024
#define LQ     (HH * WW)          // 9216
#define NTOK   (NB * LQ)          // 18432

typedef __attribute__((ext_vector_type(16))) __bf16 v16bf;
typedef __attribute__((ext_vector_type(8)))  float  v8f;
typedef __attribute__((ext_vector_type(4)))  int    v4i;

union FragB {
    v16bf v;
    unsigned int u[8];
};

#if __has_builtin(__builtin_amdgcn_global_load_async_to_lds_b128)
#define HAVE_ASYNC_LDS 1
#define LDS_AS __attribute__((address_space(3)))
#define GLB_AS __attribute__((address_space(1)))
#else
#define HAVE_ASYNC_LDS 0
#endif

static __device__ __forceinline__ void wait_async0() {
#if HAVE_ASYNC_LDS
#if __has_builtin(__builtin_amdgcn_s_wait_asynccnt)
    __builtin_amdgcn_s_wait_asynccnt(0);
#else
    asm volatile("s_wait_asynccnt 0x0" ::: "memory");
#endif
#endif
}

static __device__ __forceinline__ unsigned short f32_to_bf16(float f) {
    unsigned int u = __builtin_bit_cast(unsigned int, f);
    unsigned int lsb = (u >> 16) & 1u;
    u += 0x7fffu + lsb;                 // round-to-nearest-even
    return (unsigned short)(u >> 16);
}

// ---------------------------------------------------------------- f32 -> bf16
__global__ void cvt_bf16_kernel(const float* __restrict__ in,
                                unsigned short* __restrict__ out, int n) {
    int i = blockIdx.x * 256 + threadIdx.x;
    if (i < n) out[i] = f32_to_bf16(in[i]);
}

// ---------------------------------------------------------------- LayerNorm -> bf16
__global__ void ln_bf16_kernel(const float* __restrict__ x,
                               const float* __restrict__ g,
                               const float* __restrict__ b,
                               unsigned short* __restrict__ out) {
    int t = blockIdx.x;
    int c = threadIdx.x;
    float v = x[(size_t)t * CC + c];
    float s = v, sq = v * v;
    #pragma unroll
    for (int o = 16; o > 0; o >>= 1) {
        s  += __shfl_xor(s,  o, 32);
        sq += __shfl_xor(sq, o, 32);
    }
    __shared__ float sbuf[16];
    int lane = c & 31, wid = c >> 5;
    if (lane == 0) { sbuf[wid] = s; sbuf[8 + wid] = sq; }
    __syncthreads();
    float ts = 0.f, tq = 0.f;
    #pragma unroll
    for (int i = 0; i < 8; i++) { ts += sbuf[i]; tq += sbuf[8 + i]; }
    float mean = ts * (1.f / CC);
    float var  = tq * (1.f / CC) - mean * mean;
    float r    = rsqrtf(var + 1e-5f);
    float y    = (v - mean) * r * g[c] + b[c];
    out[(size_t)t * CC + c] = f32_to_bf16(y);
}

// ---------------------------------------------------------------- WMMA GEMM
// C[M,N] = A[M,K](bf16) @ B[K,N](bf16) + bias (+res) (gelu?) -> f32/bf16
// block 256 thr = 8 wave32 as 4(M) x 2(N); block tile 128x64; wave tile 32x32
// (4 v_wmma_f32_16x16x32_bf16 per K-step). Double-buffered LDS; A staged via
// GLOBAL_LOAD_ASYNC_TO_LDS_B128 (ASYNCcnt) when available, B via regs with
// in-register transpose-pack -> 4x ds_store_b32.
struct BPack { unsigned int w[4]; };

template <bool GELU, bool RES, bool OUTBF16>
__global__ __launch_bounds__(256)
void gemm_wmma_kernel(const unsigned short* __restrict__ A,
                      const unsigned short* __restrict__ B,
                      const float* __restrict__ bias,
                      const float* __restrict__ res,
                      void* __restrict__ outp,
                      int M, int K, int N) {
    __shared__ unsigned short As[2][128 * 40];   // [buf][row][k] row stride 40
    __shared__ unsigned short Bs[2][64 * 40];    // [buf][col][k] packed pairs

    const int mb   = blockIdx.y * 128;
    const int nb   = blockIdx.x * 64;
    const int tid  = threadIdx.x;
    const int lane = tid & 31;
    const int wid  = tid >> 5;
    const int wm   = wid >> 1;          // 0..3
    const int wn   = wid & 1;           // 0..1
    const int hi   = lane >> 4;
    const int l15  = lane & 15;

    v8f c00 = {}, c01 = {}, c10 = {}, c11 = {};

    // A staging: 2x 16B chunks per thread
    const int ar = tid >> 2;            // 0..63, second chunk row +64
    const int ac = (tid & 3) << 3;      // 0/8/16/24
    const unsigned short* Ag0 = A + (size_t)(mb + ar) * K + ac;
    const unsigned short* Ag1 = A + (size_t)(mb + ar + 64) * K + ac;
    const int al0 = ar * 40 + ac;
    const int al1 = (ar + 64) * 40 + ac;

    // B staging: k-pair kp (rows 2kp,2kp+1), col chunk cc (4 cols)
    const int kp  = tid >> 4;           // 0..15
    const int cc  = tid & 15;           // 0..15
    const int bcol = nb + cc * 4;
    const unsigned short* Bg = B + (size_t)(2 * kp) * N + bcol;

    const int NK = K >> 5;

    auto load_b = [&](int k0) {
        BPack r;
        if (bcol < N) {
            const unsigned short* p = Bg + (size_t)k0 * N;
            uint2 r0 = *(const uint2*)p;
            uint2 r1 = *(const uint2*)(p + N);
            r.w[0] = (r0.x & 0xffffu) | (r1.x << 16);
            r.w[1] = (r0.x >> 16)     | (r1.x & 0xffff0000u);
            r.w[2] = (r0.y & 0xffffu) | (r1.y << 16);
            r.w[3] = (r0.y >> 16)     | (r1.y & 0xffff0000u);
        } else {
            r.w[0] = r.w[1] = r.w[2] = r.w[3] = 0u;
        }
        return r;
    };
    auto store_b = [&](int buf, const BPack& r) {
        #pragma unroll
        for (int i = 0; i < 4; i++)
            *(unsigned int*)(&Bs[buf][(cc * 4 + i) * 40 + 2 * kp]) = r.w[i];
    };

#if HAVE_ASYNC_LDS
    auto stage_a = [&](int k0, int buf) {
        __builtin_amdgcn_global_load_async_to_lds_b128(
            (GLB_AS v4i*)(Ag0 + k0), (LDS_AS v4i*)(&As[buf][al0]), 0, 0);
        __builtin_amdgcn_global_load_async_to_lds_b128(
            (GLB_AS v4i*)(Ag1 + k0), (LDS_AS v4i*)(&As[buf][al1]), 0, 0);
    };
#else
    uint4 aq0, aq1;
    auto stage_a = [&](int k0, int /*buf*/) {
        aq0 = *(const uint4*)(Ag0 + k0);
        aq1 = *(const uint4*)(Ag1 + k0);
    };
    auto store_a = [&](int buf) {
        *(uint4*)(&As[buf][al0]) = aq0;
        *(uint4*)(&As[buf][al1]) = aq1;
    };
#endif

    // fragment row bases
    const int arow0 = (wm * 32 + l15) * 40;
    const int arow1 = (wm * 32 + 16 + l15) * 40;
    const int brow0 = (wn * 32 + l15) * 40;
    const int brow1 = (wn * 32 + 16 + l15) * 40;

    // ---- prologue: tile 0
    int cur = 0;
    stage_a(0, 0);
    BPack bp = load_b(0);
    wait_async0();
#if !HAVE_ASYNC_LDS
    store_a(0);
#endif
    store_b(0, bp);
    __syncthreads();

    for (int kt = 0; kt < NK; ++kt) {
        const bool nxt = (kt + 1) < NK;
        if (nxt) {
            stage_a((kt + 1) * 32, cur ^ 1);
            bp = load_b((kt + 1) * 32);
        }
        // ---- compute current tile
        FragB a0, a1, b0, b1;
        #pragma unroll
        for (int v = 0; v < 8; v++) {
            // A frag (ISA 16-bit A 16x32): k = (v/4)*16 + half*8 + (v%4)*2
            const int ka = ((v >> 2) << 4) + (hi << 3) + ((v & 3) << 1);
            // B frag (same K order as A): k = (v/4)*16 + half*8 + (v%4)*2
            const int kb = ((v >> 2) << 4) + (hi << 3) + ((v & 3) << 1);
            a0.u[v] = *(const unsigned int*)(&As[cur][arow0 + ka]);
            a1.u[v] = *(const unsigned int*)(&As[cur][arow1 + ka]);
            b0.u[v] = *(const unsigned int*)(&Bs[cur][brow0 + kb]);
            b1.u[v] = *(const unsigned int*)(&Bs[cur][brow1 + kb]);
        }
        c00 = __builtin_amdgcn_wmma_f32_16x16x32_bf16(false, a0.v, false, b0.v,
                                                      (short)0, c00, false, false);
        c01 = __builtin_amdgcn_wmma_f32_16x16x32_bf16(false, a0.v, false, b1.v,
                                                      (short)0, c01, false, false);
        c10 = __builtin_amdgcn_wmma_f32_16x16x32_bf16(false, a1.v, false, b0.v,
                                                      (short)0, c10, false, false);
        c11 = __builtin_amdgcn_wmma_f32_16x16x32_bf16(false, a1.v, false, b1.v,
                                                      (short)0, c11, false, false);
        if (nxt) {
            wait_async0();
            __syncthreads();              // everyone done reading buf cur
#if !HAVE_ASYNC_LDS
            store_a(cur ^ 1);
#endif
            store_b(cur ^ 1, bp);
            __syncthreads();              // next tile visible
            cur ^= 1;
        }
    }

    // ---- epilogue: D layout: vgpr v, lane half -> row = half*8+v, col = lane&15
    const v8f* acc[2][2] = { { &c00, &c01 }, { &c10, &c11 } };
    #pragma unroll
    for (int ti = 0; ti < 2; ti++) {
        #pragma unroll
        for (int tj = 0; tj < 2; tj++) {
            const v8f& cc_ = *acc[ti][tj];
            const int gn = nb + wn * 32 + tj * 16 + l15;
            if (gn < N) {
                #pragma unroll
                for (int v = 0; v < 8; v++) {
                    const int gm = mb + wm * 32 + ti * 16 + hi * 8 + v;
                    float val = cc_[v] + bias[gn];
                    if (GELU) val = 0.5f * val * (1.f + erff(val * 0.70710678118654752f));
                    if (RES)  val += res[(size_t)gm * N + gn];
                    if (OUTBF16)
                        ((unsigned short*)outp)[(size_t)gm * N + gn] = f32_to_bf16(val);
                    else
                        ((float*)outp)[(size_t)gm * N + gn] = val;
                }
            }
        }
    }
}

// ---------------------------------------------------------------- softmax over P=25
__global__ void softmax_p_kernel(float* __restrict__ logits, int rows) {
    int lane = threadIdx.x & 31;
    int row  = blockIdx.x * 8 + (threadIdx.x >> 5);
    if (row >= rows) return;
    float v = (lane < PP) ? logits[(size_t)row * PP + lane] : -3.4e38f;
    float m = v;
    #pragma unroll
    for (int o = 16; o > 0; o >>= 1) m = fmaxf(m, __shfl_xor(m, o, 32));
    float e = (lane < PP) ? expf(v - m) : 0.f;
    float s = e;
    #pragma unroll
    for (int o = 16; o > 0; o >>= 1) s += __shfl_xor(s, o, 32);
    if (lane < PP) logits[(size_t)row * PP + lane] = e / s;
}

// ---------------------------------------------------------------- deformable sampling
__global__ void deform_sample_kernel(const float* __restrict__ value,  // [NTOK][C]
                                     const float* __restrict__ offb,   // [NTOK][H*P*2]
                                     const float* __restrict__ aw,     // [NTOK][H*P]
                                     const float* __restrict__ ref,    // [NTOK][2]
                                     unsigned short* __restrict__ out) // [NTOK][C] bf16
{
    int lane  = threadIdx.x & 31;
    int gw    = blockIdx.x * 8 + (threadIdx.x >> 5);
    int token = gw >> 3;          // HEADS == 8
    int head  = gw & 7;
    int n     = token / LQ;

    float refx = ref[2 * token + 0];
    float refy = ref[2 * token + 1];
    const float* op    = offb + (size_t)token * (HEADS * PP * 2) + head * (PP * 2);
    const float* ap    = aw   + (size_t)token * (HEADS * PP)     + head * PP;
    const float* vbase = value + (size_t)n * LQ * CC + head * DHH + lane;

    float acc = 0.f;
    for (int p = 0; p < PP; p++) {
        float gx = refx * (float)WW + op[2 * p + 0] - 0.5f;
        float gy = refy * (float)HH + op[2 * p + 1] - 0.5f;
        float x0f = floorf(gx), y0f = floorf(gy);
        float fx = gx - x0f, fy = gy - y0f;
        int x0 = (int)x0f, y0 = (int)y0f;
        float a = ap[p];
        #pragma unroll
        for (int dy = 0; dy < 2; dy++) {
            #pragma unroll
            for (int dx = 0; dx < 2; dx++) {
                int xi = x0 + dx, yi = y0 + dy;
                float wgt = (dx ? fx : 1.f - fx) * (dy ? fy : 1.f - fy);
                bool valid = (xi >= 0) && (xi < WW) && (yi >= 0) && (yi < HH);
                int xc = min(max(xi, 0), WW - 1);
                int yc = min(max(yi, 0), HH - 1);
                float v = vbase[(size_t)(yc * WW + xc) * CC];
                acc += valid ? (a * wgt * v) : 0.f;
            }
        }
    }
    out[(size_t)token * CC + head * DHH + lane] = f32_to_bf16(acc);
}

// ---------------------------------------------------------------- launch
extern "C" void kernel_launch(void* const* d_in, const int* in_sizes, int n_in,
                              void* d_out, int out_size, void* d_ws, size_t ws_size,
                              hipStream_t stream) {
    const float* x      = (const float*)d_in[0];
    const float* refp   = (const float*)d_in[1];
    const float* ln1_g  = (const float*)d_in[4];
    const float* ln1_b  = (const float*)d_in[5];
    const float* w_off  = (const float*)d_in[6];
    const float* b_off  = (const float*)d_in[7];
    const float* w_attn = (const float*)d_in[8];
    const float* b_attn = (const float*)d_in[9];
    const float* w_val  = (const float*)d_in[10];
    const float* b_val  = (const float*)d_in[11];
    const float* w_out  = (const float*)d_in[12];
    const float* b_out  = (const float*)d_in[13];
    const float* ln2_g  = (const float*)d_in[14];
    const float* ln2_b  = (const float*)d_in[15];
    const float* w_fc1  = (const float*)d_in[16];
    const float* b_fc1  = (const float*)d_in[17];
    const float* w_fc2  = (const float*)d_in[18];
    const float* b_fc2  = (const float*)d_in[19];
    float* out = (float*)d_out;

    size_t cur = 0;
    auto alloc = [&](size_t bytes) -> void* {
        cur = (cur + 255) & ~(size_t)255;
        void* p = (char*)d_ws + cur;
        cur += bytes;
        return p;
    };
    unsigned short* wb_off  = (unsigned short*)alloc((size_t)CC * 400 * 2);
    unsigned short* wb_attn = (unsigned short*)alloc((size_t)CC * 200 * 2);
    unsigned short* wb_val  = (unsigned short*)alloc((size_t)CC * CC * 2);
    unsigned short* wb_out  = (unsigned short*)alloc((size_t)CC * CC * 2);
    unsigned short* wb_fc1  = (unsigned short*)alloc((size_t)CC * HIDN * 2);
    unsigned short* wb_fc2  = (unsigned short*)alloc((size_t)HIDN * CC * 2);
    unsigned short* q_bf    = (unsigned short*)alloc((size_t)NTOK * CC * 2);
    float*          valbuf  = (float*)alloc((size_t)NTOK * CC * 4);
    float*          offbuf  = (float*)alloc((size_t)NTOK * 400 * 4);
    float*          logits  = (float*)alloc((size_t)NTOK * 200 * 4);
    unsigned short* attn_bf = (unsigned short*)alloc((size_t)NTOK * CC * 2);
    float*          x1      = (float*)alloc((size_t)NTOK * CC * 4);
    unsigned short* ln2_bf  = (unsigned short*)alloc((size_t)NTOK * CC * 2);
    unsigned short* h1_bf   = (unsigned short*)alloc((size_t)NTOK * HIDN * 2);

    auto cvt = [&](const float* src, unsigned short* dst, int n) {
        cvt_bf16_kernel<<<(n + 255) / 256, 256, 0, stream>>>(src, dst, n);
    };
    cvt(w_off,  wb_off,  CC * 400);
    cvt(w_attn, wb_attn, CC * 200);
    cvt(w_val,  wb_val,  CC * CC);
    cvt(w_out,  wb_out,  CC * CC);
    cvt(w_fc1,  wb_fc1,  CC * HIDN);
    cvt(w_fc2,  wb_fc2,  HIDN * CC);

    ln_bf16_kernel<<<NTOK, 256, 0, stream>>>(x, ln1_g, ln1_b, q_bf);

    const int MT = NTOK / 128;   // 144
    gemm_wmma_kernel<false, false, false><<<dim3(CC / 64, MT), 256, 0, stream>>>(
        q_bf, wb_val, b_val, nullptr, valbuf, NTOK, CC, CC);
    gemm_wmma_kernel<false, false, false><<<dim3((400 + 63) / 64, MT), 256, 0, stream>>>(
        q_bf, wb_off, b_off, nullptr, offbuf, NTOK, CC, 400);
    gemm_wmma_kernel<false, false, false><<<dim3((200 + 63) / 64, MT), 256, 0, stream>>>(
        q_bf, wb_attn, b_attn, nullptr, logits, NTOK, CC, 200);

    softmax_p_kernel<<<(NTOK * HEADS) / 8, 256, 0, stream>>>(logits, NTOK * HEADS);

    deform_sample_kernel<<<(NTOK * HEADS) / 8, 256, 0, stream>>>(
        valbuf, offbuf, logits, refp, attn_bf);

    gemm_wmma_kernel<false, true, false><<<dim3(CC / 64, MT), 256, 0, stream>>>(
        attn_bf, wb_out, b_out, x, x1, NTOK, CC, CC);

    ln_bf16_kernel<<<NTOK, 256, 0, stream>>>(x1, ln2_g, ln2_b, ln2_bf);

    gemm_wmma_kernel<true, false, true><<<dim3(HIDN / 64, MT), 256, 0, stream>>>(
        ln2_bf, wb_fc1, b_fc1, nullptr, h1_bf, NTOK, CC, HIDN);

    gemm_wmma_kernel<false, true, false><<<dim3(CC / 64, MT), 256, 0, stream>>>(
        h1_bf, wb_fc2, b_fc2, x1, out, NTOK, HIDN, CC);

    (void)in_sizes; (void)n_in; (void)out_size; (void)ws_size;
}
